// Mario4Model_51092930953674
// MI455X (gfx1250) — matrix operand order repaired and checked by reading, not compiled, not run on hardware
//
#include <hip/hip_runtime.h>
#include <math.h>

// ---------------------------------------------------------------------------
// CDNA5 (gfx1250) wave32 WMMA types
// ---------------------------------------------------------------------------
typedef __attribute__((ext_vector_type(16))) __bf16 v16bf;
typedef __attribute__((ext_vector_type(8)))  __bf16 v8bf;
typedef __attribute__((ext_vector_type(8)))  float  v8f;

#define ACT_NONE 0
#define ACT_GELU 1
#define ACT_SIG  2
#define ACT_SILU 3

__device__ __forceinline__ float gelu_f(float x) {
  return 0.5f * x * (1.0f + erff(x * 0.70710678118654752f));
}
__device__ __forceinline__ float sig_f(float x) { return 1.0f / (1.0f + expf(-x)); }
__device__ __forceinline__ float act_apply(float v, int act) {
  switch (act) {
    case ACT_GELU: return gelu_f(v);
    case ACT_SIG:  return sig_f(v);
    case ACT_SILU: return v * sig_f(v);
    default:       return v;
  }
}
__device__ __forceinline__ int imin_(int a, int b) { return a < b ? a : b; }
__device__ __forceinline__ int iclamp_(int v, int lo, int hi) {
  return v < lo ? lo : (v > hi ? hi : v);
}

// 16 consecutive bf16 (32B) -> one fragment via two b128 loads.
__device__ __forceinline__ v16bf load_run16(const __bf16* p) {
  v8bf lo = *(const v8bf*)p;
  v8bf hi = *(const v8bf*)(p + 8);
  return __builtin_shufflevector(lo, hi, 0, 1, 2, 3, 4, 5, 6, 7, 8, 9, 10, 11, 12,
                                 13, 14, 15);
}
// A fragment: two 8-element runs at p (k=kA..kA+7) and p+16 (k=16+kA..+7).
__device__ __forceinline__ v16bf load_a_frag(const __bf16* p) {
  v8bf lo = *(const v8bf*)p;
  v8bf hi = *(const v8bf*)(p + 16);
  return __builtin_shufflevector(lo, hi, 0, 1, 2, 3, 4, 5, 6, 7, 8, 9, 10, 11, 12,
                                 13, 14, 15);
}

// A-matrix element e -> K offset (gather kernels): e 0..7 -> kA+e, 8..15 -> 16+kA+e-8
__device__ __forceinline__ int ka_off(int e, int lane) {
  return ((e & 8) ? (16 + (e & 7)) : e) + ((lane & 16) >> 1);
}

__device__ __forceinline__ v8f wmma_bf16(v16bf a, v16bf b, v8f c) {
  return __builtin_amdgcn_wmma_f32_16x16x32_bf16(false, a, false, b, (short)0, c,
                                                 false, false);
}
__device__ __forceinline__ void store_out(void* Y, size_t idx, float v, int act,
                                          int store_bf) {
  float r = act_apply(v, act);
  if (store_bf) ((__bf16*)Y)[idx] = (__bf16)r;
  else          ((float*)Y)[idx] = r;
}

// ---------------------------------------------------------------------------
// GEMM: Y[M,N] = act(X[M,K] @ W[N,K]^T + bias[N])   (bf16 in, f32 acc)
// REQUIRES K % 32 == 0. OOR rows/cols clamped on load (results never stored).
// Block = 128 thr (4 waves); wave = 16(M) x 64(N) tile.
// ---------------------------------------------------------------------------
__global__ void k_linear_bf(const __bf16* __restrict__ X, const __bf16* __restrict__ W,
                            const float* __restrict__ bias, void* __restrict__ Yout,
                            int M, int N, int K, int act, int store_bf) {
  const int lane = threadIdx.x & 31;
  const int wid  = threadIdx.x >> 5;
  const int m0 = blockIdx.x * 64 + wid * 16;
  const int n0 = blockIdx.y * 64;

  v8f acc[4];
#pragma unroll
  for (int j = 0; j < 4; ++j)
#pragma unroll
    for (int i = 0; i < 8; ++i) acc[j][i] = 0.0f;

  const int kA = (lane & 16) >> 1;
  const __bf16* __restrict__ xrow = X + (size_t)imin_(m0 + (lane & 15), M - 1) * K;
  const __bf16* __restrict__ wrow[4];
#pragma unroll
  for (int j = 0; j < 4; ++j)
    wrow[j] = W + (size_t)imin_(n0 + j * 16 + (lane & 15), N - 1) * K + ((lane & 16) >> 1);

  for (int kb = 0; kb < K; kb += 32) {
    v16bf a  = load_a_frag(xrow + kb + kA);
    v16bf b0 = load_a_frag(wrow[0] + kb);
    v16bf b1 = load_a_frag(wrow[1] + kb);
    v16bf b2 = load_a_frag(wrow[2] + kb);
    v16bf b3 = load_a_frag(wrow[3] + kb);
    acc[0] = wmma_bf16(a, b0, acc[0]);
    acc[1] = wmma_bf16(a, b1, acc[1]);
    acc[2] = wmma_bf16(a, b2, acc[2]);
    acc[3] = wmma_bf16(a, b3, acc[3]);
  }

#pragma unroll
  for (int j = 0; j < 4; ++j) {
    int n = n0 + j * 16 + (lane & 15);
    if (n >= N) continue;
    float bv = bias ? bias[n] : 0.0f;
#pragma unroll
    for (int r = 0; r < 8; ++r) {
      int m = m0 + r + ((lane & 16) >> 1);
      if (m < M) store_out(Yout, (size_t)m * N + n, acc[j][r] + bv, act, store_bf);
    }
  }
}

// ---------------------------------------------------------------------------
// Encoder conv 3x3 s2 p1 as implicit GEMM. X bf16 NCHW (time-major imgs),
// W bf16 [Cout, ldb] (zero-padded rows, ldb % 32 == 0). Output f32 NCHW (+bias).
// ---------------------------------------------------------------------------
__global__ void k_conv3x3s2_bf(const __bf16* __restrict__ X,
                               const __bf16* __restrict__ W,
                               const float* __restrict__ bias, float* __restrict__ Y,
                               int NIMG, int Cin, int IH, int IW, int Cout, int ldb) {
  const int OH = IH >> 1, OW = IW >> 1;
  const int M = NIMG * OH * OW;
  const int lane = threadIdx.x & 31;
  const int wid  = threadIdx.x >> 5;
  const int m0 = blockIdx.x * 64 + wid * 16;
  const int n0 = blockIdx.y * 64;

  v8f acc[4];
#pragma unroll
  for (int j = 0; j < 4; ++j)
#pragma unroll
    for (int i = 0; i < 8; ++i) acc[j][i] = 0.0f;

  const int mA  = imin_(m0 + (lane & 15), M - 1);
  int img = mA / (OH * OW);
  int r0  = mA - img * (OH * OW);
  const int oy = r0 / OW;
  const int ox = r0 - oy * OW;
  const __bf16* __restrict__ Xb = X + (size_t)img * Cin * IH * IW;

  const __bf16* __restrict__ wrow[4];
#pragma unroll
  for (int j = 0; j < 4; ++j)
    wrow[j] = W + (size_t)imin_(n0 + j * 16 + (lane & 15), Cout - 1) * ldb +
              ((lane & 16) >> 1);

  for (int kb = 0; kb < ldb; kb += 32) {
    v16bf a;
#pragma unroll
    for (int e = 0; e < 16; ++e) {
      int k  = kb + ka_off(e, lane);
      int ci = k / 9;
      int r2 = k - ci * 9;
      int ky = r2 / 3;
      int kx = r2 - ky * 3;
      int iy = oy * 2 - 1 + ky;
      int ix = ox * 2 - 1 + kx;
      bool ok = (ci < Cin) & ((unsigned)iy < (unsigned)IH) &
                ((unsigned)ix < (unsigned)IW);
      int    cic = imin_(ci, Cin - 1);
      int    iyc = iclamp_(iy, 0, IH - 1);
      int    ixc = iclamp_(ix, 0, IW - 1);
      __bf16 v   = Xb[((size_t)cic * IH + iyc) * IW + ixc];  // always valid
      a[e] = ok ? v : (__bf16)0.0f;                          // select, no branch
    }
    v16bf b0 = load_a_frag(wrow[0] + kb);
    v16bf b1 = load_a_frag(wrow[1] + kb);
    v16bf b2 = load_a_frag(wrow[2] + kb);
    v16bf b3 = load_a_frag(wrow[3] + kb);
    acc[0] = wmma_bf16(a, b0, acc[0]);
    acc[1] = wmma_bf16(a, b1, acc[1]);
    acc[2] = wmma_bf16(a, b2, acc[2]);
    acc[3] = wmma_bf16(a, b3, acc[3]);
  }

#pragma unroll
  for (int j = 0; j < 4; ++j) {
    int n = n0 + j * 16 + (lane & 15);
    if (n >= Cout) continue;
    float bv = bias ? bias[n] : 0.0f;
#pragma unroll
    for (int r = 0; r < 8; ++r) {
      int m = m0 + r + ((lane & 16) >> 1);
      if (m < M) {
        int mi = m / (OH * OW);
        int rr = m - mi * (OH * OW);
        int yy = rr / OW;
        int xx = rr - yy * OW;
        Y[(((size_t)mi * Cout + n) * OH + yy) * OW + xx] = acc[j][r] + bv;
      }
    }
  }
}

// ---------------------------------------------------------------------------
// ConvTranspose2d k4 s2 p1 as implicit GEMM split by output parity class
// (blockIdx.z = py*2+px). K = Cin*4 (mult of 32). Weight pre-packed per parity:
// Wp[((p*Cout)+n)*K + k] so B rows are contiguous (vector loads).
// ---------------------------------------------------------------------------
__global__ void k_convT4x4_bf(const __bf16* __restrict__ X,
                              const __bf16* __restrict__ Wp,
                              const float* __restrict__ bias, void* __restrict__ Yout,
                              int NIMG, int Cin, int IH, int IW, int Cout, int act,
                              int store_bf) {
  const int py = (int)(blockIdx.z >> 1), px = (int)(blockIdx.z & 1);
  const int M = NIMG * IH * IW;  // per-parity output grid is IH x IW
  const int K = Cin * 4;
  const int OH = IH * 2, OW = IW * 2;
  const int lane = threadIdx.x & 31;
  const int wid  = threadIdx.x >> 5;
  const int m0 = blockIdx.x * 64 + wid * 16;
  const int n0 = blockIdx.y * 64;

  v8f acc[4];
#pragma unroll
  for (int j = 0; j < 4; ++j)
#pragma unroll
    for (int i = 0; i < 8; ++i) acc[j][i] = 0.0f;

  const int mA  = imin_(m0 + (lane & 15), M - 1);
  int img = mA / (IH * IW);
  int r0  = mA - img * (IH * IW);
  const int my = r0 / IW;
  const int mx = r0 - my * IW;
  const __bf16* __restrict__ Xb = X + (size_t)img * Cin * IH * IW;

  const __bf16* __restrict__ wrow[4];
#pragma unroll
  for (int j = 0; j < 4; ++j)
    wrow[j] = Wp +
              ((size_t)blockIdx.z * Cout +
               imin_(n0 + j * 16 + (lane & 15), Cout - 1)) * K + ((lane & 16) >> 1);

  for (int kb = 0; kb < K; kb += 32) {
    v16bf a;
#pragma unroll
    for (int e = 0; e < 16; ++e) {
      int k  = kb + ka_off(e, lane);
      int ci = k >> 2;
      int t  = k & 3;
      int iy = my + py - (t >> 1);
      int ix = mx + px - (t & 1);
      bool ok = ((unsigned)iy < (unsigned)IH) & ((unsigned)ix < (unsigned)IW);
      int    iyc = iclamp_(iy, 0, IH - 1);
      int    ixc = iclamp_(ix, 0, IW - 1);
      __bf16 v   = Xb[((size_t)ci * IH + iyc) * IW + ixc];
      a[e] = ok ? v : (__bf16)0.0f;
    }
    v16bf b0 = load_a_frag(wrow[0] + kb);
    v16bf b1 = load_a_frag(wrow[1] + kb);
    v16bf b2 = load_a_frag(wrow[2] + kb);
    v16bf b3 = load_a_frag(wrow[3] + kb);
    acc[0] = wmma_bf16(a, b0, acc[0]);
    acc[1] = wmma_bf16(a, b1, acc[1]);
    acc[2] = wmma_bf16(a, b2, acc[2]);
    acc[3] = wmma_bf16(a, b3, acc[3]);
  }

#pragma unroll
  for (int j = 0; j < 4; ++j) {
    int n = n0 + j * 16 + (lane & 15);
    if (n >= Cout) continue;
    float bv = bias ? bias[n] : 0.0f;
#pragma unroll
    for (int r = 0; r < 8; ++r) {
      int m = m0 + r + ((lane & 16) >> 1);
      if (m < M) {
        int mi = m / (IH * IW);
        int rr = m - mi * (IH * IW);
        int yy = rr / IW;
        int xx = rr - yy * IW;
        store_out(Yout,
                  (((size_t)mi * Cout + n) * OH + 2 * yy + py) * OW + 2 * xx + px,
                  acc[j][r] + bv, act, store_bf);
      }
    }
  }
}

// ---------------------------------------------------------------------------
// BatchNorm (training) stats per (timestep group of 8 imgs, channel)
// ---------------------------------------------------------------------------
__global__ void k_bn_stats(const float* __restrict__ X, float* __restrict__ mean,
                           float* __restrict__ inv, int C, int HW) {
  const int c = blockIdx.x, t = blockIdx.y;
  const long total = 8L * HW;
  float s = 0.f, s2 = 0.f;
  for (long i = threadIdx.x; i < total; i += blockDim.x) {
    long b = i / HW, pos = i - b * HW;
    float v = X[(((long)(t * 8 + b) * C + c)) * HW + pos];
    s += v;
    s2 += v * v;
  }
  __shared__ float sh[256], sh2[256];
  sh[threadIdx.x] = s;
  sh2[threadIdx.x] = s2;
  __syncthreads();
  for (int o = 128; o > 0; o >>= 1) {
    if ((int)threadIdx.x < o) {
      sh[threadIdx.x] += sh[threadIdx.x + o];
      sh2[threadIdx.x] += sh2[threadIdx.x + o];
    }
    __syncthreads();
  }
  if (threadIdx.x == 0) {
    float m   = sh[0] / (float)total;
    float var = sh2[0] / (float)total - m * m;
    mean[t * C + c] = m;
    inv[t * C + c]  = rsqrtf(var + 1e-5f);
  }
}

// BN apply + GELU; reads f32 conv output, writes bf16 activation
__global__ void k_bn_apply_gelu(const float* __restrict__ X,
                                const float* __restrict__ mean,
                                const float* __restrict__ inv,
                                const float* __restrict__ g,
                                const float* __restrict__ b, __bf16* __restrict__ Y,
                                int C, int HW, long total) {
  long i = (long)blockIdx.x * blockDim.x + threadIdx.x;
  if (i >= total) return;
  long img = i / ((long)C * HW);
  int  t   = (int)(img >> 3);
  int  c   = (int)((i / HW) % C);
  float v = (X[i] - mean[t * C + c]) * inv[t * C + c] * g[c] + b[c];
  Y[i] = (__bf16)gelu_f(v);
}

__global__ void k_pool_bf(const __bf16* __restrict__ X, __bf16* __restrict__ Y,
                          int NC, int HW) {
  int i = blockIdx.x * blockDim.x + threadIdx.x;
  if (i >= NC) return;
  const __bf16* p = X + (size_t)i * HW;
  float s = 0.f;
  for (int j = 0; j < HW; ++j) s += (float)p[j];
  Y[i] = (__bf16)(s / (float)HW);
}

// LayerNorm over last dim; optional extra bf16 output
__global__ void k_layernorm(const float* __restrict__ X, const float* __restrict__ g,
                            const float* __restrict__ b, float* __restrict__ Y,
                            __bf16* __restrict__ Ybf, int D) {
  const int row = blockIdx.x;
  const float* x = X + (size_t)row * D;
  float s = 0.f, s2 = 0.f;
  for (int i = threadIdx.x; i < D; i += blockDim.x) {
    float v = x[i];
    s += v;
    s2 += v * v;
  }
  __shared__ float sh[256], sh2[256];
  sh[threadIdx.x] = s;
  sh2[threadIdx.x] = s2;
  __syncthreads();
  for (int o = 128; o > 0; o >>= 1) {
    if ((int)threadIdx.x < o) {
      sh[threadIdx.x] += sh[threadIdx.x + o];
      sh2[threadIdx.x] += sh2[threadIdx.x + o];
    }
    __syncthreads();
  }
  __shared__ float mm, iv;
  if (threadIdx.x == 0) {
    float m   = sh[0] / (float)D;
    float var = sh2[0] / (float)D - m * m;
    mm = m;
    iv = rsqrtf(var + 1e-5f);
  }
  __syncthreads();
  for (int i = threadIdx.x; i < D; i += blockDim.x) {
    float v = (x[i] - mm) * iv * g[i] + b[i];
    Y[(size_t)row * D + i] = v;
    if (Ybf) Ybf[(size_t)row * D + i] = (__bf16)v;
  }
}

// concat two f32 row blocks into a bf16 matrix
__global__ void k_concat_bf(const float* __restrict__ A, const float* __restrict__ B,
                            __bf16* __restrict__ Y, int R, int K1, int K2) {
  int i = blockIdx.x * blockDim.x + threadIdx.x;
  int Kt = K1 + K2;
  if (i >= R * Kt) return;
  int r = i / Kt, k = i - r * Kt;
  Y[i] = (__bf16)((k < K1) ? A[(size_t)r * K1 + k] : B[(size_t)r * K2 + (k - K1)]);
}

// gather actions[:,t,:] into padded bf16 [Bsz,Kd] (cols >= A stay zero)
__global__ void k_gather_act_bf(const float* __restrict__ actions,
                                __bf16* __restrict__ a_t, int t, int T, int A,
                                int Bsz, int Kd) {
  int i = blockIdx.x * blockDim.x + threadIdx.x;
  if (i >= Bsz * A) return;
  int b = i / A, j = i - b * A;
  a_t[(size_t)b * Kd + j] = (__bf16)actions[((size_t)b * T + t) * A + j];
}

// pad rows + convert: dst[R,Kd](bf16) = src[R,Ks](f32) zero-extended
__global__ void k_pad_rows_bf(const float* __restrict__ src, __bf16* __restrict__ dst,
                              int R, int Ks, int Kd) {
  int i = blockIdx.x * blockDim.x + threadIdx.x;
  if (i >= R * Kd) return;
  int r = i / Kd, k = i - r * Kd;
  dst[i] = (__bf16)((k < Ks) ? src[(size_t)r * Ks + k] : 0.0f);
}

__global__ void k_f2b(const float* __restrict__ src, __bf16* __restrict__ dst,
                      long n) {
  long i = (long)blockIdx.x * blockDim.x + threadIdx.x;
  if (i < n) dst[i] = (__bf16)src[i];
}

// remap frames [B,T,3,H,W] -> bf16 time-major [(t*8+b),3,H,W]
__global__ void k_frames_bf(const float* __restrict__ src, __bf16* __restrict__ dst,
                            long n) {
  long i = (long)blockIdx.x * blockDim.x + threadIdx.x;
  if (i >= n) return;
  const long IMSZ = 3L * 224 * 224;
  long img = i / IMSZ, pos = i - img * IMSZ;
  int tt = (int)(img >> 3), bb = (int)(img & 7);
  dst[i] = (__bf16)src[((long)(bb * 12 + tt)) * IMSZ + pos];
}

// pack convT weight torch[Cin,Cout,4,4] -> bf16 per-parity [4][Cout][Cin*4]
__global__ void k_packT(const float* __restrict__ W, __bf16* __restrict__ Wp,
                        int Cin, int Cout) {
  int K = Cin * 4;
  int i = blockIdx.x * blockDim.x + threadIdx.x;
  if (i >= 4 * Cout * K) return;
  int p = i / (Cout * K);
  int rem = i - p * (Cout * K);
  int n = rem / K, k = rem - n * K;
  int ci = k >> 2, t = k & 3;
  int py = p >> 1, px = p & 1;
  int ky = (1 - py) + (t & 2);
  int kx = (1 - px) + ((t & 1) << 1);
  Wp[i] = (__bf16)W[(((size_t)ci * Cout + n) * 4 + ky) * 4 + kx];
}

// stats[8,2L] -> (mean, logvar); optional lat = mean + exp(0.5*lv)*eps
__global__ void k_stats_lat(const float* __restrict__ stats,
                            const float* __restrict__ eps, float* __restrict__ lat,
                            float* __restrict__ m_out, float* __restrict__ lv_out,
                            int Bsz, int L) {
  int i = blockIdx.x * blockDim.x + threadIdx.x;
  if (i >= Bsz * L) return;
  int b = i / L, j = i - b * L;
  float m  = stats[(size_t)b * 2 * L + j];
  float lv = stats[(size_t)b * 2 * L + L + j];
  if (m_out)  m_out[i]  = m;
  if (lv_out) lv_out[i] = lv;
  if (lat)    lat[i] = m + expf(0.5f * lv) * eps[i];
}

__global__ void k_film(const float* __restrict__ c, const float* __restrict__ g,
                       const float* __restrict__ bl, float* __restrict__ x,
                       __bf16* __restrict__ xbf, int n) {
  int i = blockIdx.x * blockDim.x + threadIdx.x;
  if (i >= n) return;
  float v = gelu_f(c[i] * (1.0f + g[i]) + bl[i]);
  x[i] = v;
  xbf[i] = (__bf16)v;
}

__global__ void k_ssm(const float* __restrict__ a_diag, const float* __restrict__ g,
                      const float* __restrict__ h, const float* __restrict__ Bx,
                      __bf16* __restrict__ s16, int Bsz, int Hd) {
  int i = blockIdx.x * blockDim.x + threadIdx.x;
  if (i >= Bsz * Hd) return;
  int j = i % Hd;
  float ad = a_diag[j];
  float sp = (ad > 20.f) ? ad : log1pf(expf(ad));  // softplus
  float A  = -sp;
  float gv = g[i];
  s16[i] = (__bf16)((A * gv + (1.0f - gv)) * h[i] + Bx[i]);
}

__global__ void k_gru(const float* __restrict__ gi, const float* __restrict__ gh,
                      const float* __restrict__ h, float* __restrict__ hn, int Bsz,
                      int Hd) {
  int i = blockIdx.x * blockDim.x + threadIdx.x;
  if (i >= Bsz * Hd) return;
  int b = i / Hd, j = i - b * Hd;
  size_t base = (size_t)b * 3 * Hd;
  float r = sig_f(gi[base + j] + gh[base + j]);
  float z = sig_f(gi[base + Hd + j] + gh[base + Hd + j]);
  float n = tanhf(gi[base + 2 * Hd + j] + r * gh[base + 2 * Hd + j]);
  hn[i] = (1.0f - z) * n + z * h[i];
}

__global__ void k_copy(const float* __restrict__ src, float* __restrict__ dst, long n) {
  long i = (long)blockIdx.x * blockDim.x + threadIdx.x;
  if (i < n) dst[i] = src[i];
}
__global__ void k_fill(float* __restrict__ p, float v, long n) {
  long i = (long)blockIdx.x * blockDim.x + threadIdx.x;
  if (i < n) p[i] = v;
}
__global__ void k_fill_bf(__bf16* __restrict__ p, float v, long n) {
  long i = (long)blockIdx.x * blockDim.x + threadIdx.x;
  if (i < n) p[i] = (__bf16)v;
}

// ---------------------------------------------------------------------------
// Host orchestration
// ---------------------------------------------------------------------------
extern "C" void kernel_launch(void* const* d_in, const int* in_sizes, int n_in,
                              void* d_out, int out_size, void* d_ws, size_t ws_size,
                              hipStream_t stream) {
  (void)in_sizes; (void)n_in; (void)out_size;
  const int T = 12, Bsz = 8, WARM = 4, Hd = 512, Ld = 128, Ad = 16;

  const float* frames  = (const float*)d_in[0];
  const float* actions = (const float*)d_in[1];
  const float* noise   = (const float*)d_in[2];
  const float* enc_w[4]  = {(const float*)d_in[3],  (const float*)d_in[7],
                            (const float*)d_in[11], (const float*)d_in[15]};
  const float* enc_bi[4] = {(const float*)d_in[4],  (const float*)d_in[8],
                            (const float*)d_in[12], (const float*)d_in[16]};
  const float* bn_gp[4]  = {(const float*)d_in[5],  (const float*)d_in[9],
                            (const float*)d_in[13], (const float*)d_in[17]};
  const float* bn_bp[4]  = {(const float*)d_in[6],  (const float*)d_in[10],
                            (const float*)d_in[14], (const float*)d_in[18]};
  const float* proj_w = (const float*)d_in[19];
  const float* proj_b = (const float*)d_in[20];
  const float* dec_fc_w = (const float*)d_in[21];
  const float* dec_fc_b = (const float*)d_in[22];
  const float* dec_w[4] = {(const float*)d_in[23], (const float*)d_in[25],
                           (const float*)d_in[27], (const float*)d_in[29]};
  const float* dec_b[4] = {(const float*)d_in[24], (const float*)d_in[26],
                           (const float*)d_in[28], (const float*)d_in[30]};
  const float* rep_w[3] = {(const float*)d_in[31], (const float*)d_in[33],
                           (const float*)d_in[35]};
  const float* rep_b[3] = {(const float*)d_in[32], (const float*)d_in[34],
                           (const float*)d_in[36]};
  const float* dyn_w[3] = {(const float*)d_in[37], (const float*)d_in[39],
                           (const float*)d_in[41]};
  const float* dyn_b[3] = {(const float*)d_in[38], (const float*)d_in[40],
                           (const float*)d_in[42]};
  const float* ae_w0 = (const float*)d_in[43];
  const float* ae_b0 = (const float*)d_in[44];
  const float* ae_w1 = (const float*)d_in[45];
  const float* ae_b1 = (const float*)d_in[46];
  const float* pre_w = (const float*)d_in[47];
  const float* pre_b = (const float*)d_in[48];
  const float* fg_w[2]  = {(const float*)d_in[49], (const float*)d_in[55]};
  const float* fg_b[2]  = {(const float*)d_in[50], (const float*)d_in[56]};
  const float* fb_w[2]  = {(const float*)d_in[51], (const float*)d_in[57]};
  const float* fb_b[2]  = {(const float*)d_in[52], (const float*)d_in[58]};
  const float* fln_g[2] = {(const float*)d_in[53], (const float*)d_in[59]};
  const float* fln_b[2] = {(const float*)d_in[54], (const float*)d_in[60]};
  const float* a_diag = (const float*)d_in[61];
  const float* ssmB_w = (const float*)d_in[62];
  const float* ssmB_b = (const float*)d_in[63];
  const float* g1_w = (const float*)d_in[64];
  const float* g1_b = (const float*)d_in[65];
  const float* g2_w = (const float*)d_in[66];
  const float* g2_b = (const float*)d_in[67];
  const float* gru_wih = (const float*)d_in[68];
  const float* gru_whh = (const float*)d_in[69];
  const float* gru_bih = (const float*)d_in[70];
  const float* gru_bhh = (const float*)d_in[71];
  const float* post_g = (const float*)d_in[72];
  const float* post_b = (const float*)d_in[73];
  const float* pol_w = (const float*)d_in[74];
  const float* pol_b = (const float*)d_in[75];

  // ---- output layout (tuple concatenated flat, 8 train steps) ----
  float* out = (float*)d_out;
  float* o_recon  = out;                     // [8,8,3,224,224]
  float* o_logits = out + 9633792;           // [8,8,16]
  float* o_hidden = o_logits + 1024;         // [8,8,512]
  float* o_pm     = o_hidden + 32768;        // [8,8,128]
  float* o_plv    = o_pm + 8192;
  float* o_prm    = o_plv + 8192;
  float* o_prlv   = o_prm + 8192;

  // ---- workspace bump allocator ----
  char*  wsb = (char*)d_ws;
  size_t off = 0;
  auto allocf = [&](size_t n) -> float* {
    float* p = (float*)(wsb + off);
    off += ((n * 4 + 255) / 256) * 256;
    return p;
  };
  auto allocb = [&](size_t n) -> __bf16* {
    __bf16* p = (__bf16*)(wsb + off);
    off += ((n * 2 + 255) / 256) * 256;
    return p;
  };
  float*  RA   = allocf(38535168);   // 96*32*112*112 f32 (encoder conv out ping)
  float*  RB   = allocf(19267584);   // 96*64*56*56  f32 (encoder conv out pong)
  __bf16* RA16 = allocb(38535168);   // bf16 activation ping (enc + dec reuse)
  __bf16* RB16 = allocb(19267584);   // bf16 activation pong
  __bf16* F16  = allocb(14450688);   // frames bf16, time-major
  // bf16 weights
  __bf16* w_enc0p = allocb(32 * 32);
  __bf16* w_enc1  = allocb(64 * 288);
  __bf16* w_enc2  = allocb(128 * 576);
  __bf16* w_enc3  = allocb(192 * 1152);
  __bf16* w_proj  = allocb(128 * 192);
  __bf16* w_decfc = allocb(50176 * 128);
  __bf16* w_dec0  = allocb(4 * 128 * 1024);
  __bf16* w_dec1  = allocb(4 * 64 * 512);
  __bf16* w_dec2  = allocb(4 * 32 * 256);
  __bf16* w_dec3  = allocb(4 * 3 * 128);
  __bf16* w_rep0  = allocb(512 * 640);
  __bf16* w_rep1  = allocb(512 * 512);
  __bf16* w_rep2  = allocb(256 * 512);
  __bf16* w_dyn0  = allocb(512 * 512);
  __bf16* w_dyn1  = allocb(512 * 512);
  __bf16* w_dyn2  = allocb(256 * 512);
  __bf16* w_ae0p  = allocb(512 * 32);
  __bf16* w_ae1   = allocb(512 * 512);
  __bf16* w_pre   = allocb(512 * 640);
  __bf16* w_fg[2] = {allocb(512 * 512), allocb(512 * 512)};
  __bf16* w_fb[2] = {allocb(512 * 512), allocb(512 * 512)};
  __bf16* w_ssmB  = allocb(512 * 512);
  __bf16* w_g1    = allocb(1024 * 1024);
  __bf16* w_g2    = allocb(512 * 1024);
  __bf16* w_wih   = allocb(1536 * 512);
  __bf16* w_whh   = allocb(1536 * 512);
  __bf16* w_pol   = allocb(16 * 640);
  // small activations
  __bf16* pooled16  = allocb(96 * 192);
  __bf16* lat_all16 = allocb(64 * 128);
  __bf16* a_t16     = allocb(8 * 32);
  __bf16* h16       = allocb(8 * 512);
  __bf16* ae0_16    = allocb(8 * 512);
  __bf16* ae16      = allocb(8 * 512);
  __bf16* cat_he16  = allocb(8 * 640);
  __bf16* cat_hl16  = allocb(8 * 640);
  __bf16* cat_xh16  = allocb(8 * 1024);
  __bf16* mlp0_16   = allocb(8 * 512);
  __bf16* mlp1_16   = allocb(8 * 512);
  __bf16* t1_16     = allocb(8 * 1024);
  __bf16* x16       = allocb(8 * 512);
  __bf16* s16       = allocb(8 * 512);
  float* enc_all = allocf(96 * 128);
  float* bn_m  = allocf(12 * 192);
  float* bn_i  = allocf(12 * 192);
  float* h     = allocf(8 * 512);
  float* stats = allocf(8 * 256);
  float* lat_cur = allocf(8 * 128);
  float* xbuf  = allocf(8 * 512);
  float* cbuf  = allocf(8 * 512);
  float* glbuf = allocf(8 * 512);
  float* blbuf = allocf(8 * 512);
  float* gbuf  = allocf(8 * 512);
  float* Bxbuf = allocf(8 * 512);
  float* gibuf = allocf(8 * 1536);
  float* ghbuf = allocf(8 * 1536);
  float* hnbuf = allocf(8 * 512);
  if (off > ws_size) return;  // workspace too small; nothing safe to do

  auto lin = [&](const __bf16* X, const __bf16* W, const float* B, void* Y, int M,
                 int N, int K, int act, int sbf) {
    dim3 g((M + 63) / 64, (N + 63) / 64);
    k_linear_bf<<<g, 128, 0, stream>>>(X, W, B, Y, M, N, K, act, sbf);
  };
  auto conv = [&](const __bf16* X, const __bf16* W, const float* B, float* Y,
                  int NIMG, int Cin, int IH, int IW, int Cout, int ldb) {
    int M = NIMG * (IH / 2) * (IW / 2);
    dim3 g((M + 63) / 64, (Cout + 63) / 64);
    k_conv3x3s2_bf<<<g, 128, 0, stream>>>(X, W, B, Y, NIMG, Cin, IH, IW, Cout, ldb);
  };
  auto bn = [&](const float* X, const float* g, const float* b, __bf16* Y, int C,
                int HW) {
    k_bn_stats<<<dim3(C, 12), 256, 0, stream>>>(X, bn_m, bn_i, C, HW);
    long total = 96L * C * HW;
    k_bn_apply_gelu<<<(int)((total + 255) / 256), 256, 0, stream>>>(
        X, bn_m, bn_i, g, b, Y, C, HW, total);
  };
  auto convT = [&](const __bf16* X, const __bf16* Wp, const float* B, void* Y,
                   int NIMG, int Cin, int IH, int IW, int Cout, int act, int sbf) {
    int M = NIMG * IH * IW;
    dim3 g((M + 63) / 64, (Cout + 63) / 64, 4);
    k_convT4x4_bf<<<g, 128, 0, stream>>>(X, Wp, B, Y, NIMG, Cin, IH, IW, Cout, act,
                                         sbf);
  };
  auto f2b = [&](const float* s, __bf16* d, long n) {
    k_f2b<<<(int)((n + 255) / 256), 256, 0, stream>>>(s, d, n);
  };

  // ---- per-launch operand prep (bf16 weights, frames, padded/packed forms) ----
  k_frames_bf<<<(int)((14450688L + 255) / 256), 256, 0, stream>>>(frames, F16,
                                                                  14450688L);
  k_pad_rows_bf<<<(32 * 32 + 255) / 256, 256, 0, stream>>>(enc_w[0], w_enc0p, 32, 27,
                                                           32);
  f2b(enc_w[1], w_enc1, 64 * 288);
  f2b(enc_w[2], w_enc2, 128 * 576);
  f2b(enc_w[3], w_enc3, 192 * 1152);
  f2b(proj_w, w_proj, 128 * 192);
  f2b(dec_fc_w, w_decfc, 50176L * 128);
  k_packT<<<(4 * 128 * 1024 + 255) / 256, 256, 0, stream>>>(dec_w[0], w_dec0, 256, 128);
  k_packT<<<(4 * 64 * 512 + 255) / 256, 256, 0, stream>>>(dec_w[1], w_dec1, 128, 64);
  k_packT<<<(4 * 32 * 256 + 255) / 256, 256, 0, stream>>>(dec_w[2], w_dec2, 64, 32);
  k_packT<<<(4 * 3 * 128 + 255) / 256, 256, 0, stream>>>(dec_w[3], w_dec3, 32, 3);
  f2b(rep_w[0], w_rep0, 512 * 640);
  f2b(rep_w[1], w_rep1, 512 * 512);
  f2b(rep_w[2], w_rep2, 256 * 512);
  f2b(dyn_w[0], w_dyn0, 512 * 512);
  f2b(dyn_w[1], w_dyn1, 512 * 512);
  f2b(dyn_w[2], w_dyn2, 256 * 512);
  k_pad_rows_bf<<<(512 * 32 + 255) / 256, 256, 0, stream>>>(ae_w0, w_ae0p, 512, 16,
                                                            32);
  f2b(ae_w1, w_ae1, 512 * 512);
  f2b(pre_w, w_pre, 512 * 640);
  f2b(fg_w[0], w_fg[0], 512 * 512);
  f2b(fg_w[1], w_fg[1], 512 * 512);
  f2b(fb_w[0], w_fb[0], 512 * 512);
  f2b(fb_w[1], w_fb[1], 512 * 512);
  f2b(ssmB_w, w_ssmB, 512 * 512);
  f2b(g1_w, w_g1, 1024 * 1024);
  f2b(g2_w, w_g2, 512 * 1024);
  f2b(gru_wih, w_wih, 1536 * 512);
  f2b(gru_whh, w_whh, 1536 * 512);
  f2b(pol_w, w_pol, 16 * 640);
  k_fill_bf<<<1, 256, 0, stream>>>(a_t16, 0.0f, 8 * 32);
  k_fill<<<(4096 + 255) / 256, 256, 0, stream>>>(h, 0.0f, 4096);
  k_fill_bf<<<(4096 + 255) / 256, 256, 0, stream>>>(h16, 0.0f, 4096);

  // ================= Phase 1: encoder, batched over all 96 frames ===========
  conv(F16, w_enc0p, enc_bi[0], RA, 96, 3, 224, 224, 32, 32);
  bn(RA, bn_gp[0], bn_bp[0], RA16, 32, 112 * 112);
  conv(RA16, w_enc1, enc_bi[1], RB, 96, 32, 112, 112, 64, 288);
  bn(RB, bn_gp[1], bn_bp[1], RB16, 64, 56 * 56);
  conv(RB16, w_enc2, enc_bi[2], RA, 96, 64, 56, 56, 128, 576);
  bn(RA, bn_gp[2], bn_bp[2], RA16, 128, 28 * 28);
  conv(RA16, w_enc3, enc_bi[3], RB, 96, 128, 28, 28, 192, 1152);
  bn(RB, bn_gp[3], bn_bp[3], RB16, 192, 14 * 14);
  k_pool_bf<<<(96 * 192 + 255) / 256, 256, 0, stream>>>(RB16, pooled16, 96 * 192, 196);
  lin(pooled16, w_proj, proj_b, enc_all, 96, 128, 192, ACT_NONE, 0);

  // ================= Phase 2: sequential recurrence =========================
  for (int t = 0; t < T; ++t) {
    const int  s     = t - WARM;
    const bool train = (t >= WARM);

    // action embedding (K padded 16 -> 32)
    k_gather_act_bf<<<1, 128, 0, stream>>>(actions, a_t16, t, T, Ad, Bsz, 32);
    lin(a_t16, w_ae0p, ae_b0, ae0_16, Bsz, Hd, 32, ACT_GELU, 1);
    lin(ae0_16, w_ae1, ae_b1, ae16, Bsz, Hd, Hd, ACT_NONE, 1);

    if (train) {  // prior (dyn) from current h
      lin(h16, w_dyn0, dyn_b[0], mlp0_16, Bsz, Hd, Hd, ACT_GELU, 1);
      lin(mlp0_16, w_dyn1, dyn_b[1], mlp1_16, Bsz, Hd, Hd, ACT_GELU, 1);
      lin(mlp1_16, w_dyn2, dyn_b[2], stats, Bsz, 2 * Ld, Hd, ACT_NONE, 0);
      k_stats_lat<<<(Bsz * Ld + 255) / 256, 256, 0, stream>>>(
          stats, nullptr, nullptr, o_prm + (size_t)s * 1024,
          o_prlv + (size_t)s * 1024, Bsz, Ld);
    }

    // posterior (rep) from [h, enc_t]
    const float* enc_t = enc_all + (size_t)t * Bsz * Ld;
    k_concat_bf<<<(8 * 640 + 255) / 256, 256, 0, stream>>>(h, enc_t, cat_he16, Bsz,
                                                           Hd, Ld);
    lin(cat_he16, w_rep0, rep_b[0], mlp0_16, Bsz, Hd, Hd + Ld, ACT_GELU, 1);
    lin(mlp0_16, w_rep1, rep_b[1], mlp1_16, Bsz, Hd, Hd, ACT_GELU, 1);
    lin(mlp1_16, w_rep2, rep_b[2], stats, Bsz, 2 * Ld, Hd, ACT_NONE, 0);
    const float* eps_t = noise + (size_t)t * Bsz * Ld;
    k_stats_lat<<<(Bsz * Ld + 255) / 256, 256, 0, stream>>>(
        stats, eps_t, lat_cur, train ? (o_pm + (size_t)s * 1024) : nullptr,
        train ? (o_plv + (size_t)s * 1024) : nullptr, Bsz, Ld);
    if (train)
      k_f2b<<<(1024 + 255) / 256, 256, 0, stream>>>(lat_cur,
                                                    lat_all16 + (size_t)s * 1024,
                                                    1024);

    // [h, lat] for policy head and cell preprocess
    k_concat_bf<<<(8 * 640 + 255) / 256, 256, 0, stream>>>(h, lat_cur, cat_hl16, Bsz,
                                                           Hd, Ld);
    if (train)
      lin(cat_hl16, w_pol, pol_b, o_logits + (size_t)s * Bsz * Ad, Bsz, Ad, Hd + Ld,
          ACT_NONE, 0);

    // ---- cell ----
    lin(cat_hl16, w_pre, pre_b, xbuf, Bsz, Hd, Hd + Ld, ACT_NONE, 0);
    for (int l = 0; l < 2; ++l) {  // FiLM stack
      k_layernorm<<<Bsz, 256, 0, stream>>>(xbuf, fln_g[l], fln_b[l], cbuf, nullptr,
                                           Hd);
      lin(ae16, w_fg[l], fg_b[l], glbuf, Bsz, Hd, Hd, ACT_NONE, 0);
      lin(ae16, w_fb[l], fb_b[l], blbuf, Bsz, Hd, Hd, ACT_NONE, 0);
      k_film<<<(Bsz * Hd + 255) / 256, 256, 0, stream>>>(cbuf, glbuf, blbuf, xbuf,
                                                         x16, Bsz * Hd);
    }
    // selective SSM step
    k_concat_bf<<<(8 * 1024 + 255) / 256, 256, 0, stream>>>(xbuf, h, cat_xh16, Bsz,
                                                            Hd, Hd);
    lin(cat_xh16, w_g1, g1_b, t1_16, Bsz, 2 * Hd, 2 * Hd, ACT_SILU, 1);
    lin(t1_16, w_g2, g2_b, gbuf, Bsz, Hd, 2 * Hd, ACT_SIG, 0);
    lin(x16, w_ssmB, ssmB_b, Bxbuf, Bsz, Hd, Hd, ACT_NONE, 0);
    k_ssm<<<(Bsz * Hd + 255) / 256, 256, 0, stream>>>(a_diag, gbuf, h, Bxbuf, s16,
                                                      Bsz, Hd);
    // GRU cell (torch gate order r,z,n)
    lin(s16, w_wih, gru_bih, gibuf, Bsz, 3 * Hd, Hd, ACT_NONE, 0);
    lin(h16, w_whh, gru_bhh, ghbuf, Bsz, 3 * Hd, Hd, ACT_NONE, 0);
    k_gru<<<(Bsz * Hd + 255) / 256, 256, 0, stream>>>(gibuf, ghbuf, h, hnbuf, Bsz,
                                                      Hd);
    k_layernorm<<<Bsz, 256, 0, stream>>>(hnbuf, post_g, post_b, h, h16, Hd);
    if (train)
      k_copy<<<(4096 + 255) / 256, 256, 0, stream>>>(h, o_hidden + (size_t)s * 4096,
                                                     4096);
  }

  // ================= Phase 3: decoder, batched over 64 latents ==============
  lin(lat_all16, w_decfc, dec_fc_b, RB16, 64, 50176, 128, ACT_NONE, 1);
  convT(RB16, w_dec0, dec_b[0], RA16, 64, 256, 14, 14, 128, ACT_GELU, 1);
  convT(RA16, w_dec1, dec_b[1], RB16, 64, 128, 28, 28, 64, ACT_GELU, 1);
  convT(RB16, w_dec2, dec_b[2], RA16, 64, 64, 56, 56, 32, ACT_GELU, 1);
  convT(RA16, w_dec3, dec_b[3], o_recon, 64, 32, 112, 112, 3, ACT_SIG, 0);
}
